// OptimizedAttention_14353780703562
// MI455X (gfx1250) — matrix operand order repaired and checked by reading, compiled, not run on hardware
//
#include <hip/hip_runtime.h>

// ---------------------------------------------------------------------------
// Fused causal attention block for MI455X (gfx1250, wave32, WMMA).
// Pipeline: f32->f16 convert | QKV GEMM (WMMA) | RoPE+RMSNorm | V transpose |
//           flash attention (WMMA QK^T + WMMA PV, online softmax) |
//           output GEMM (WMMA) -> f32.
// LDS staging is double-buffered; uses GLOBAL_LOAD_ASYNC_TO_LDS (ASYNCcnt)
// when the toolchain exposes the builtin, else a synchronous VGPR round-trip.
// ---------------------------------------------------------------------------

typedef __attribute__((ext_vector_type(16))) _Float16 v16h;
typedef __attribute__((ext_vector_type(8)))  _Float16 v8h;
typedef __attribute__((ext_vector_type(8)))  float    v8f;
typedef __attribute__((ext_vector_type(4)))  int      v4i;

union F16frag { v16h v; v8h h8[2]; uint4 q[2]; };
union H4 { _Float16 h[4]; uint2 u; };

#define WMMA_F16(a, b, c) \
  __builtin_amdgcn_wmma_f32_16x16x32_f16(false, (a), false, (b), (short)0, (c), false, false)

#if defined(__has_builtin)
#if __has_builtin(__builtin_amdgcn_global_load_async_to_lds_b128)
#define HAVE_ASYNC_LDS 1
#endif
#endif

typedef __attribute__((address_space(1))) v4i* gptr_v4i;
typedef __attribute__((address_space(3))) v4i* sptr_v4i;

// Copy 16 bytes global -> LDS (async when available).
__device__ __forceinline__ void cp16_g2s(_Float16* lds, const _Float16* g) {
#ifdef HAVE_ASYNC_LDS
  __builtin_amdgcn_global_load_async_to_lds_b128(
      (gptr_v4i)(const_cast<_Float16*>(g)), (sptr_v4i)(lds), 0, 0);
#else
  *reinterpret_cast<uint4*>(lds) = *reinterpret_cast<const uint4*>(g);
#endif
}

__device__ __forceinline__ void cp_wait() {
#ifdef HAVE_ASYNC_LDS
#if __has_builtin(__builtin_amdgcn_s_wait_asynccnt)
  __builtin_amdgcn_s_wait_asynccnt(0);
#else
  asm volatile("s_wait_asynccnt 0" ::: "memory");
#endif
#endif
}

// ---------------------------------------------------------------------------
// Kernel 1: f32 -> f16 conversion (grid-stride, float4 granularity)
// ---------------------------------------------------------------------------
__global__ void fa_cvt_f32_f16(const float* __restrict__ s,
                               _Float16* __restrict__ d, long n4) {
  long stride = (long)gridDim.x * blockDim.x;
  for (long i = (long)blockIdx.x * blockDim.x + threadIdx.x; i < n4; i += stride) {
    float4 v = *reinterpret_cast<const float4*>(s + i * 4);
    H4 o;
    o.h[0] = (_Float16)v.x; o.h[1] = (_Float16)v.y;
    o.h[2] = (_Float16)v.z; o.h[3] = (_Float16)v.w;
    *reinterpret_cast<uint2*>(d + i * 4) = o.u;
  }
}

// ---------------------------------------------------------------------------
// Kernel 2: GEMM  C[M,N] = A[M,K] * B[N,K]^T   (A,B f16 row-major K-contig,
// C f32). Block = 8 waves; wave computes 32(M)x64(N); B tile double-buffered
// in LDS with async global->LDS staging; one barrier per k-step.
// ---------------------------------------------------------------------------
__global__ __launch_bounds__(256) void fa_gemm_f16f32(
    const _Float16* __restrict__ A, const _Float16* __restrict__ B,
    float* __restrict__ C, int M, int N, int K) {
  __shared__ _Float16 bs[2][64 * 40];  // 64 N-rows x 32 K halves, stride 40
  const int tid  = threadIdx.x;
  const int lane = tid & 31;
  const int wave = tid >> 5;
  const int half = lane >> 4;
  const int l16  = lane & 15;
  const int n0 = blockIdx.x * 64;
  const int m0 = blockIdx.y * 256 + wave * 32;
  const int srow = tid >> 2, sseg = tid & 3;  // staging: one 16B chunk/thread

  v8f acc[2][4];
#pragma unroll
  for (int mi = 0; mi < 2; ++mi)
#pragma unroll
    for (int nd = 0; nd < 4; ++nd) { v8f z = {}; acc[mi][nd] = z; }

  // prologue: stage k-tile 0 into buffer 0
  cp16_g2s(&bs[0][srow * 40 + sseg * 8],
           B + (size_t)(n0 + srow) * K + sseg * 8);
  cp_wait();
  __syncthreads();

  for (int kb = 0, it = 0; kb < K; kb += 32, ++it) {
    const int cur = it & 1;
    if (kb + 32 < K)  // prefetch next tile into the other buffer
      cp16_g2s(&bs[cur ^ 1][srow * 40 + sseg * 8],
               B + (size_t)(n0 + srow) * K + kb + 32 + sseg * 8);

    F16frag bf[4];
#pragma unroll
    for (int nd = 0; nd < 4; ++nd) {
      // B layout: lane col n = l16; lanes<16 K{0..7,16..23}, lanes>=16 K{8..15,24..31}
      const uint4* p = reinterpret_cast<const uint4*>(
          &bs[cur][(nd * 16 + l16) * 40]);
      bf[nd].q[0] = p[half];
      bf[nd].q[1] = p[2 + half];
    }
#pragma unroll
    for (int mi = 0; mi < 2; ++mi) {
      // A layout: row m = l16; lanes<16 K{0..7,16..23}, lanes>=16 K{8..15,24..31}
      const _Float16* arow = A + (size_t)(m0 + mi * 16 + l16) * K + kb;
      F16frag af;
      af.q[0] = *reinterpret_cast<const uint4*>(arow + half * 8);
      af.q[1] = *reinterpret_cast<const uint4*>(arow + 16 + half * 8);
#pragma unroll
      for (int nd = 0; nd < 4; ++nd)
        acc[mi][nd] = WMMA_F16(af.v, bf[nd].v, acc[mi][nd]);
    }

    if (kb + 32 < K) {
      cp_wait();        // our async writes to the next buffer have landed
      __syncthreads();  // everyone done reading cur + everyone's writes done
    }
  }

#pragma unroll
  for (int mi = 0; mi < 2; ++mi)
#pragma unroll
    for (int nd = 0; nd < 4; ++nd)
#pragma unroll
      for (int r = 0; r < 8; ++r) {
        int row = m0 + mi * 16 + r + 8 * half;   // C layout: row r + 8*(lane/16)
        int col = n0 + nd * 16 + l16;
        C[(size_t)row * N + col] = acc[mi][nd][r];
      }
}

// ---------------------------------------------------------------------------
// Kernel 3: RoPE + RMS norm for q and k. One wave per (b,h,t) row (D=128),
// lane covers 4 elements (2 rotation pairs). Emits f16 [B*H, T, 128].
// ---------------------------------------------------------------------------
__global__ __launch_bounds__(256) void fa_rope_rms(
    const float* __restrict__ qkv,   // [B, T, 3, H, 128]
    _Float16* __restrict__ Qh, _Float16* __restrict__ Kh,
    int Bn, int Tt, int H) {
  const int wave = threadIdx.x >> 5, lane = threadIdx.x & 31;
  const long row = (long)blockIdx.x * 8 + wave;           // over B*H*T
  const int t  = (int)(row % Tt);
  const int bh = (int)(row / Tt);
  const int h = bh % H, b = bh / H;
  const int D = 128;
  const float lnf = 9.2103403719761836f;  // ln(10000)

  const float inv0 = __expf(-((float)(4 * lane)     / 128.f) * lnf);
  const float inv1 = __expf(-((float)(4 * lane + 2) / 128.f) * lnf);
  float s0, c0, s1, c1;
  __sincosf((float)t * inv0, &s0, &c0);
  __sincosf((float)t * inv1, &s1, &c1);

#pragma unroll
  for (int s = 0; s < 2; ++s) {  // 0 = q, 1 = k
    const float* src = qkv + (size_t)(b * Tt + t) * (3 * H * D) +
                       (size_t)s * H * D + h * D + lane * 4;
    float4 u = *reinterpret_cast<const float4*>(src);
    float o0 = u.x * c0 - u.y * s0;
    float o1 = u.y * c0 + u.x * s0;
    float o2 = u.z * c1 - u.w * s1;
    float o3 = u.w * c1 + u.z * s1;
    float ss = o0 * o0 + o1 * o1 + o2 * o2 + o3 * o3;
#pragma unroll
    for (int m = 1; m < 32; m <<= 1) ss += __shfl_xor(ss, m, 32);
    float rms = rsqrtf(ss * (1.0f / 128.0f) + 1.1920929e-07f);
    H4 o;
    o.h[0] = (_Float16)(o0 * rms); o.h[1] = (_Float16)(o1 * rms);
    o.h[2] = (_Float16)(o2 * rms); o.h[3] = (_Float16)(o3 * rms);
    _Float16* dst = (s == 0 ? Qh : Kh) + ((size_t)bh * Tt + t) * D + lane * 4;
    *reinterpret_cast<uint2*>(dst) = o.u;
  }
}

// ---------------------------------------------------------------------------
// Kernel 4: V transpose: qkv v-slice [B,T,H,128] f32 -> Vt [B*H, 128, T] f16
// via an LDS tile (32 t x 128 d), coalesced both ways.
// ---------------------------------------------------------------------------
__global__ __launch_bounds__(256) void fa_vtrans(
    const float* __restrict__ qkv, _Float16* __restrict__ Vt,
    int Bn, int Tt, int H) {
  __shared__ _Float16 tile[128 * 40];  // [d][t] padded
  const int bh = blockIdx.y, b = bh / H, h = bh % H;
  const int t0 = blockIdx.x * 32;
  const int tid = threadIdx.x;
  const int D = 128;
#pragma unroll
  for (int i = 0; i < 16; ++i) {
    int idx = i * 256 + tid;           // 0..4095
    int r = idx >> 7, c = idx & 127;   // r = local t, c = d
    float v = qkv[(size_t)(b * Tt + t0 + r) * (3 * H * D) +
                  (size_t)2 * H * D + h * D + c];
    tile[c * 40 + r] = (_Float16)v;
  }
  __syncthreads();
  {
    int d = tid >> 1, seg = tid & 1;
    const uint4* s = reinterpret_cast<const uint4*>(&tile[d * 40 + seg * 16]);
    uint4* dst = reinterpret_cast<uint4*>(
        Vt + ((size_t)bh * D + d) * Tt + t0 + seg * 16);
    dst[0] = s[0];
    dst[1] = s[1];
  }
}

// ---------------------------------------------------------------------------
// Kernel 5: flash attention. Block = (b, h, 128-query tile), 8 waves x 16 q.
// K/V 32-key tiles double-buffered in LDS (async staging), shared by all
// waves; S and PV via WMMA; online softmax with 16-lane shuffles; P converted
// C->A layout through wave-private LDS.
// ---------------------------------------------------------------------------
__global__ __launch_bounds__(256) void fa_attn(
    const _Float16* __restrict__ Q,   // [B*H, T, 128]
    const _Float16* __restrict__ Km,  // [B*H, T, 128]
    const _Float16* __restrict__ Vt,  // [B*H, 128, T]
    _Float16* __restrict__ O,         // [B, T, H*128] f16
    int Tt, int H) {
  const int DH = 128;
  __shared__ _Float16 ks[2][32 * 136];   // keys x d, padded
  __shared__ _Float16 vs[2][128 * 40];   // d x keys, padded
  __shared__ _Float16 pls[8][16 * 32];

  const int tid = threadIdx.x, lane = tid & 31, wave = tid >> 5;
  const int half = lane >> 4, l16 = lane & 15;
  const int qt = blockIdx.x, h = blockIdx.y, b = blockIdx.z;
  const int bh = b * H + h;
  const int q0 = qt * 128 + wave * 16;
  const size_t baseQ = (size_t)bh * Tt * DH;
  const size_t baseV = (size_t)bh * DH * Tt;
  const float scale = 0.08838834764831845f;  // 1/sqrt(128)

  F16frag qf[4];
  {
    const _Float16* qrow = Q + baseQ + (size_t)(q0 + l16) * DH;
#pragma unroll
    for (int c = 0; c < 4; ++c) {
      qf[c].q[0] = *reinterpret_cast<const uint4*>(qrow + c * 32 + half * 8);
      qf[c].q[1] = *reinterpret_cast<const uint4*>(qrow + c * 32 + 16 + half * 8);
    }
  }

  v8f o_acc[8];
#pragma unroll
  for (int nd = 0; nd < 8; ++nd) { v8f z = {}; o_acc[nd] = z; }
  float m_i[8], l_i[8];
#pragma unroll
  for (int r = 0; r < 8; ++r) { m_i[r] = -1e30f; l_i[r] = 0.f; }

  // staging helper indices (one 16B chunk per thread per step)
  const int krow = tid >> 4, kseg = tid & 15;   // 32 x 16 chunks, x2 steps
  const int vrow = tid >> 2, vseg = tid & 3;    // 128 x 4 chunks, x2 steps

  const int kend = (qt + 1) * 128;

  // prologue: stage kv-tile 0 into buffer 0
  {
#pragma unroll
    for (int i = 0; i < 2; ++i) {
      int r = krow + i * 16, sgn = kseg;
      cp16_g2s(&ks[0][r * 136 + sgn * 8],
               Km + baseQ + (size_t)r * DH + sgn * 8);
    }
#pragma unroll
    for (int i = 0; i < 2; ++i) {
      int r = vrow + i * 64, sgn = vseg;
      cp16_g2s(&vs[0][r * 40 + sgn * 8],
               Vt + baseV + (size_t)r * Tt + sgn * 8);
    }
    cp_wait();
    __syncthreads();
  }

  for (int kb = 0, it = 0; kb < kend; kb += 32, ++it) {
    const int cur = it & 1;
    if (kb + 32 < kend) {  // prefetch next kv tile into the other buffer
#pragma unroll
      for (int i = 0; i < 2; ++i) {
        int r = krow + i * 16;
        cp16_g2s(&ks[cur ^ 1][r * 136 + kseg * 8],
                 Km + baseQ + (size_t)(kb + 32 + r) * DH + kseg * 8);
      }
#pragma unroll
      for (int i = 0; i < 2; ++i) {
        int r = vrow + i * 64;
        cp16_g2s(&vs[cur ^ 1][r * 40 + vseg * 8],
                 Vt + baseV + (size_t)r * Tt + kb + 32 + vseg * 8);
      }
    }

    if (kb <= q0 + 15) {
      // S = q . K^T  (2 key sub-tiles of 16, contraction over D in 4 chunks)
      v8f s[2];
      { v8f z = {}; s[0] = z; s[1] = z; }
#pragma unroll
      for (int n = 0; n < 2; ++n)
#pragma unroll
        for (int c = 0; c < 4; ++c) {
          F16frag kf;
          const uint4* p = reinterpret_cast<const uint4*>(
              &ks[cur][(n * 16 + l16) * 136 + c * 32]);
          kf.q[0] = p[half];
          kf.q[1] = p[2 + half];
          s[n] = WMMA_F16(qf[c].v, kf.v, s[n]);
        }

      // online softmax, per row m = r + 8*half
#pragma unroll
      for (int r = 0; r < 8; ++r) {
        const int qrow = q0 + r + 8 * half;
        float mx = -1e30f;
#pragma unroll
        for (int n = 0; n < 2; ++n) {
          float v = s[n][r] * scale;
          if (kb + n * 16 + l16 > qrow) v = -1e30f;  // causal mask
          s[n][r] = v;
          mx = fmaxf(mx, v);
        }
#pragma unroll
        for (int m = 1; m < 16; m <<= 1) mx = fmaxf(mx, __shfl_xor(mx, m, 32));
        const float mnew = fmaxf(m_i[r], mx);
        const float alpha = __expf(m_i[r] - mnew);
        m_i[r] = mnew;
        float rsum = 0.f;
#pragma unroll
        for (int n = 0; n < 2; ++n) {
          float p = __expf(s[n][r] - mnew);
          s[n][r] = p;
          rsum += p;
        }
#pragma unroll
        for (int m = 1; m < 16; m <<= 1) rsum += __shfl_xor(rsum, m, 32);
        l_i[r] = l_i[r] * alpha + rsum;
#pragma unroll
        for (int nd = 0; nd < 8; ++nd) o_acc[nd][r] *= alpha;
      }

      // P: C-layout -> A-layout via wave-private LDS
      _Float16* pw = &pls[wave][0];
#pragma unroll
      for (int r = 0; r < 8; ++r)
#pragma unroll
        for (int n = 0; n < 2; ++n)
          pw[(r + 8 * half) * 32 + n * 16 + l16] = (_Float16)s[n][r];
      asm volatile("s_wait_dscnt 0" ::: "memory");

      F16frag pf;
      {
        const _Float16* pr = pw + l16 * 32;
        pf.q[0] = *reinterpret_cast<const uint4*>(pr + half * 8);
        pf.q[1] = *reinterpret_cast<const uint4*>(pr + 16 + half * 8);
      }
      // O += P . V  (8 output tiles of 16 d-cols, contraction over 32 keys)
#pragma unroll
      for (int nd = 0; nd < 8; ++nd) {
        F16frag vf;
        const uint4* p = reinterpret_cast<const uint4*>(
            &vs[cur][(nd * 16 + l16) * 40]);
        vf.q[0] = p[half];
        vf.q[1] = p[2 + half];
        o_acc[nd] = WMMA_F16(pf.v, vf.v, o_acc[nd]);
      }
    }

    if (kb + 32 < kend) {
      cp_wait();
      __syncthreads();
    }
  }

  // epilogue: normalize and scatter to [B, T, C] (head-interleaved)
  const int C = H * DH;
#pragma unroll
  for (int r = 0; r < 8; ++r) {
    const float inv = 1.0f / l_i[r];
    const int trow = q0 + r + 8 * half;
#pragma unroll
    for (int nd = 0; nd < 8; ++nd) {
      const int col = h * DH + nd * 16 + l16;
      O[((size_t)b * Tt + trow) * C + col] = (_Float16)(o_acc[nd][r] * inv);
    }
  }
}

// ---------------------------------------------------------------------------
// Launch
// ---------------------------------------------------------------------------
extern "C" void kernel_launch(void* const* d_in, const int* in_sizes, int n_in,
                              void* d_out, int out_size, void* d_ws, size_t ws_size,
                              hipStream_t stream) {
  constexpr int Bn = 2, T = 2048, C = 2048, H = 16, D = 128;
  const float* x    = (const float*)d_in[0];  // [B,T,C]
  const float* Wqkv = (const float*)d_in[1];  // [3C,C]
  const float* Wo   = (const float*)d_in[2];  // [C,C]
  float* out = (float*)d_out;                 // [B,T,C]

  char* ws = (char*)d_ws;
  size_t off = 0;
  _Float16* xh    = (_Float16*)(ws + off); off += (size_t)Bn * T * C * 2;       // 16 MB
  _Float16* wqkvh = (_Float16*)(ws + off); off += (size_t)3 * C * C * 2;        // 24 MB
  _Float16* woh   = (_Float16*)(ws + off); off += (size_t)C * C * 2;            //  8 MB
  float*    qkvf  = (float*)   (ws + off); off += (size_t)Bn * T * 3 * C * 4;   // 96 MB
  _Float16* qh    = (_Float16*)(ws + off); off += (size_t)Bn * H * T * D * 2;   // 16 MB
  _Float16* kh    = (_Float16*)(ws + off); off += (size_t)Bn * H * T * D * 2;   // 16 MB
  _Float16* vth   = (_Float16*)(ws + off); off += (size_t)Bn * H * D * T * 2;   // 16 MB
  _Float16* oh    = (_Float16*)(ws + off); off += (size_t)Bn * T * C * 2;       // 16 MB

  // 1) convert inputs to f16
  fa_cvt_f32_f16<<<1024, 256, 0, stream>>>(x,    xh,    (long)Bn * T * C / 4);
  fa_cvt_f32_f16<<<1024, 256, 0, stream>>>(Wqkv, wqkvh, (long)3 * C * C / 4);
  fa_cvt_f32_f16<<<1024, 256, 0, stream>>>(Wo,   woh,   (long)C * C / 4);

  // 2) QKV projection: [4096,2048] x [6144,2048]^T -> f32 qkv
  fa_gemm_f16f32<<<dim3(3 * C / 64, Bn * T / 256), 256, 0, stream>>>(
      xh, wqkvh, qkvf, Bn * T, 3 * C, C);

  // 3) RoPE + RMS norm -> f16 q,k in [B*H,T,D]
  fa_rope_rms<<<(Bn * H * T) / 8, 256, 0, stream>>>(qkvf, qh, kh, Bn, T, H);

  // 4) V transpose -> f16 [B*H,D,T]
  fa_vtrans<<<dim3(T / 32, Bn * H), 256, 0, stream>>>(qkvf, vth, Bn, T, H);

  // 5) causal flash attention -> f16 [B,T,C]
  fa_attn<<<dim3(T / 128, H, Bn), 256, 0, stream>>>(qh, kh, vth, oh, T, H);

  // 6) output projection: [4096,2048] x [2048,2048]^T -> f32 out
  fa_gemm_f16f32<<<dim3(C / 64, Bn * T / 256), 256, 0, stream>>>(
      oh, woh, out, Bn * T, C, C);
}
